// RelativeCrossAttentionModule_11184094839025
// MI455X (gfx1250) — matrix operand order repaired and checked by reading, not compiled, not run on hardware
//
#include <hip/hip_runtime.h>
#include <hip/hip_bf16.h>

typedef _Float16 h16;
typedef __attribute__((ext_vector_type(16))) _Float16 v16h;
typedef __attribute__((ext_vector_type(8)))  _Float16 v8h;
typedef __attribute__((ext_vector_type(8)))  float    v8f;

#define E_DIM  256
#define H_NUM  8
#define HD_DIM 32
#define LQ     512
#define BATCH  4
#define LKV    4096
#define NL     4
#define NQ     (LQ * BATCH)    /* 2048  */
#define NKV    (LKV * BATCH)   /* 16384 */

// ---------------------------------------------------------------------------
// WMMA helper: D = A(16x32 f16) x B(32x16 f16) + C(16x16 f32)
// ---------------------------------------------------------------------------
__device__ __forceinline__ v8f wmma_f16(v16h a, v16h b, v8f c) {
  return __builtin_amdgcn_wmma_f32_16x16x32_f16(
      /*neg_a=*/false, a, /*neg_b=*/false, b,
      /*c_mod=*/(short)0, c, /*reuse_a=*/false, /*reuse_b=*/false);
}

__device__ __forceinline__ v16h cat8(v8h lo, v8h hi) {
  return __builtin_shufflevector(lo, hi, 0, 1, 2, 3, 4, 5, 6, 7,
                                 8, 9, 10, 11, 12, 13, 14, 15);
}

// Async global->LDS copy of 16B per lane (CDNA5 async-tensor path, ASYNCcnt).
__device__ __forceinline__ void async_copy_b128(unsigned lds_off,
                                                unsigned long long gaddr) {
  asm volatile("global_load_async_to_lds_b128 %0, %1, off"
               :: "v"(lds_off), "v"(gaddr) : "memory");
}
__device__ __forceinline__ void wait_asynccnt0() {
  asm volatile("s_wait_asynccnt 0x0" ::: "memory");
}

// ---------------------------------------------------------------------------
// fp32 -> fp16 conversion (weights, once per launch)
// ---------------------------------------------------------------------------
__global__ void cvt_f32_f16(const float* __restrict__ src,
                            h16* __restrict__ dst, int n) {
  int i = blockIdx.x * 256 + threadIdx.x;
  if (i < n) dst[i] = (h16)src[i];
}

// ---------------------------------------------------------------------------
// Generic WMMA GEMM:  Y[M,256] = epi( X[M,256] @ Wh[256,256]^T + bias )
// Wh row-major [n, k] (f16).  Block = 128 thr (4 waves), 16 rows per block,
// wave w computes n-tiles 4w..4w+3.  Epilogue: (+bias)*scale, relu?, +resid?
// ---------------------------------------------------------------------------
__global__ void __launch_bounds__(128)
gemm16(const float* __restrict__ X, const h16* __restrict__ Wh,
       const float* __restrict__ bias, const float* __restrict__ resid,
       float* __restrict__ Y, float scale, int do_relu) {
  __shared__ __attribute__((aligned(32))) h16 ldsX[16 * E_DIM];  // 8 KB

  const int tid  = threadIdx.x;
  const int wave = tid >> 5;
  const int lane = tid & 31;
  const int l16  = lane & 15;
  const int khalf = lane >> 4;
  const int mrow0 = blockIdx.x * 16;

  // stage X tile (16 x 256 f32 -> f16)
  const float* xsrc = X + (size_t)mrow0 * E_DIM;
#pragma unroll 4
  for (int i = tid; i < 16 * E_DIM; i += 128) ldsX[i] = (h16)xsrc[i];
  __syncthreads();

  v8f acc[4] = {v8f{}, v8f{}, v8f{}, v8f{}};

#pragma unroll 2
  for (int kc = 0; kc < 8; ++kc) {
    // A fragment: row = l16, K = kc*32 + khalf*16 + e   (contiguous)
    const h16* abase = ldsX + l16 * E_DIM + kc * 32 + khalf * 16;
    v16h a = *(const v16h*)abase;
#pragma unroll
    for (int nt = 0; nt < 4; ++nt) {
      const int ncol = (wave * 4 + nt) * 16 + l16;
      // B fragment: col = ncol, K = kc*32 + khalf*16 + e   (contiguous)
      const h16* bbase = Wh + (size_t)ncol * E_DIM + kc * 32 + khalf * 16;
      v16h b = *(const v16h*)bbase;
      acc[nt] = wmma_f16(a, b, acc[nt]);
    }
  }

#pragma unroll
  for (int nt = 0; nt < 4; ++nt) {
    const int ncol = (wave * 4 + nt) * 16 + l16;
    const float bv = bias[ncol];
#pragma unroll
    for (int r = 0; r < 8; ++r) {
      const int row = mrow0 + r + khalf * 8;
      float val = (acc[nt][r] + bv) * scale;
      if (do_relu) val = fmaxf(val, 0.0f);
      if (resid) val += resid[(size_t)row * E_DIM + ncol];
      Y[(size_t)row * E_DIM + ncol] = val;
    }
  }
}

// ---------------------------------------------------------------------------
// Rotary embed + head-major reorder (f32 [L,B,E] -> f16 [B,H,L,HD]).
// One block per token, thread = feature pair.
// ---------------------------------------------------------------------------
__global__ void __launch_bounds__(128)
rotary_heads(const float* __restrict__ pre, const float* __restrict__ pos,
             h16* __restrict__ dst, int L, int do_rot) {
  const int t = blockIdx.x;       // token = l*B + b
  const int e = threadIdx.x;      // pair index 0..127
  const int l = t / BATCH, b = t % BATCH;
  const int e0 = 2 * e, e1 = 2 * e + 1;
  const float x0 = pre[(size_t)t * E_DIM + e0];
  const float x1 = pre[(size_t)t * E_DIM + e1];
  float y0 = x0, y1 = x1;
  if (do_rot) {
    const float* pp = pos + ((size_t)(b * L + l) * E_DIM) * 2;
    y0 = x0 * pp[e0 * 2 + 0] - x1 * pp[e0 * 2 + 1];
    y1 = x1 * pp[e1 * 2 + 0] + x0 * pp[e1 * 2 + 1];
  }
  const int h = e0 >> 5, hd = e0 & 31;
  h16* d = dst + ((size_t)(b * H_NUM + h) * L + l) * HD_DIM + hd;
  d[0] = (h16)y0;
  d[1] = (h16)y1;
}

// ---------------------------------------------------------------------------
// Flash attention v2.  Block (128 thr, 4 waves) owns one (b,h) pair and four
// 16-row q tiles (wave w -> tile group*4+w).  K/V streamed in 64-key chunks:
//   - K chunk (4 KB) staged raw [key][hd] via global_load_async_to_lds_b128
//   - V chunk staged transposed Vt[hd][key] cooperatively (VALU scatter)
//   - per chunk, per wave: 4 score WMMAs + online softmax + 4 P@V WMMAs
// ---------------------------------------------------------------------------
__global__ void __launch_bounds__(128)
attn_fa(const h16* __restrict__ qh, const h16* __restrict__ kh,
        const h16* __restrict__ vh, float* __restrict__ o) {
  __shared__ __attribute__((aligned(32))) h16 ldsK[64 * 32];      // 4 KB raw K
  __shared__ __attribute__((aligned(32))) h16 ldsVt[32 * 64];     // 4 KB V^T
  __shared__ __attribute__((aligned(32))) h16 ldsP[4][16 * 64];   // 4 x 2 KB

  const int tid  = threadIdx.x;
  const int wave = tid >> 5;
  const int lane = tid & 31;
  const int l16  = lane & 15;
  const int khalf = lane >> 4;

  const int bh = blockIdx.x >> 3;          // 0..31  (b*8 + h)
  const int qt = (blockIdx.x & 7) * 4 + wave;  // q tile 0..31
  const int b  = bh >> 3, h = bh & 7;

  const h16* Q = qh + ((size_t)bh * LQ + qt * 16) * HD_DIM;
  const h16* K = kh + (size_t)bh * LKV * HD_DIM;
  const h16* V = vh + (size_t)bh * LKV * HD_DIM;

  // Q A-fragment (persistent): row = l16, hd = khalf*16 + e   (contiguous)
  const h16* qa = Q + l16 * HD_DIM + khalf * 16;
  const v16h qfrag = *(const v16h*)qa;

  v8f acc0 = {}, acc1 = {};
  float m[8], lsum[8];
#pragma unroll
  for (int r = 0; r < 8; ++r) { m[r] = -1e30f; lsum[r] = 0.0f; }

  h16* P = ldsP[wave];
  const unsigned ldsK_base = (unsigned)(uintptr_t)(&ldsK[0]);
  const int vkey = tid >> 1;       // 0..63 : key row this thread stages
  const int vhalf = tid & 1;       // 0/1   : hd 0..15 or 16..31

  for (int c = 0; c < LKV / 64; ++c) {
    const int kbase = c * 64;

    __syncthreads();  // previous chunk fully consumed by all waves

    // ---- stage K chunk: 4 KB, 32 B per thread via async-to-LDS ------------
    {
      const unsigned long long gK =
          (unsigned long long)(uintptr_t)(K + (size_t)kbase * HD_DIM) + tid * 32u;
      async_copy_b128(ldsK_base + tid * 32u, gK);
      async_copy_b128(ldsK_base + tid * 32u + 16u, gK + 16u);
    }

    // ---- stage V chunk transposed: Vt[hd][key] ----------------------------
    {
      const h16* vr = V + (size_t)(kbase + vkey) * HD_DIM + vhalf * 16;
      v16h vv = *(const v16h*)vr;
#pragma unroll
      for (int d0 = 0; d0 < 16; ++d0)
        ldsVt[(vhalf * 16 + d0) * 64 + vkey] = vv[d0];
    }

    wait_asynccnt0();   // K async transfers done (this wave's lanes)
    __syncthreads();    // all staging visible block-wide

    // ---- scores: S = Q @ K^T  (four 16-key tiles) -------------------------
    v8f s0, s1, s2, s3;
    {
      const h16* kb0 = ldsK + (0 * 16 + l16) * 32 + khalf * 16;
      const h16* kb1 = ldsK + (1 * 16 + l16) * 32 + khalf * 16;
      const h16* kb2 = ldsK + (2 * 16 + l16) * 32 + khalf * 16;
      const h16* kb3 = ldsK + (3 * 16 + l16) * 32 + khalf * 16;
      s0 = wmma_f16(qfrag, *(const v16h*)kb0, v8f{});
      s1 = wmma_f16(qfrag, *(const v16h*)kb1, v8f{});
      s2 = wmma_f16(qfrag, *(const v16h*)kb2, v8f{});
      s3 = wmma_f16(qfrag, *(const v16h*)kb3, v8f{});
    }

    // ---- online softmax update (one per 64 keys) --------------------------
#pragma unroll
    for (int r = 0; r < 8; ++r) {
      float t = fmaxf(fmaxf(s0[r], s1[r]), fmaxf(s2[r], s3[r]));
      t = fmaxf(t, __shfl_xor(t, 1));
      t = fmaxf(t, __shfl_xor(t, 2));
      t = fmaxf(t, __shfl_xor(t, 4));
      t = fmaxf(t, __shfl_xor(t, 8));
      const float mn = fmaxf(m[r], t);
      const float p0 = __expf(s0[r] - mn);
      const float p1 = __expf(s1[r] - mn);
      const float p2 = __expf(s2[r] - mn);
      const float p3 = __expf(s3[r] - mn);
      float rs = (p0 + p1) + (p2 + p3);
      rs += __shfl_xor(rs, 1);
      rs += __shfl_xor(rs, 2);
      rs += __shfl_xor(rs, 4);
      rs += __shfl_xor(rs, 8);
      const float corr = __expf(m[r] - mn);
      lsum[r] = lsum[r] * corr + rs;
      m[r] = mn;
      acc0[r] *= corr;
      acc1[r] *= corr;
      const int row = r + khalf * 8;
      P[row * 64 + 0 * 16 + l16] = (h16)p0;
      P[row * 64 + 1 * 16 + l16] = (h16)p1;
      P[row * 64 + 2 * 16 + l16] = (h16)p2;
      P[row * 64 + 3 * 16 + l16] = (h16)p3;
    }

    // ---- O += P @ V  (two 32-key sub-chunks) ------------------------------
#pragma unroll
    for (int kc = 0; kc < 2; ++kc) {
      const h16* pa = P + l16 * 64 + kc * 32 + khalf * 16;
      v16h pfrag = *(const v16h*)pa;
      const h16* vb0 = ldsVt + (0 + l16) * 64 + kc * 32 + khalf * 16;
      const h16* vb1 = ldsVt + (16 + l16) * 64 + kc * 32 + khalf * 16;
      acc0 = wmma_f16(pfrag, *(const v16h*)vb0, acc0);
      acc1 = wmma_f16(pfrag, *(const v16h*)vb1, acc1);
    }
  }

  // ---- finalize: o[l,b, h*32+col] = acc/lsum ------------------------------
#pragma unroll
  for (int r = 0; r < 8; ++r) {
    const int qrow = qt * 16 + r + khalf * 8;
    const float inv = 1.0f / lsum[r];
    const size_t tok = (size_t)qrow * BATCH + b;
    o[tok * E_DIM + h * 32 + l16]      = acc0[r] * inv;
    o[tok * E_DIM + h * 32 + 16 + l16] = acc1[r] * inv;
  }
}

// ---------------------------------------------------------------------------
// LayerNorm over E=256, one wave per token (8 elems/lane), block = 4 waves.
// ---------------------------------------------------------------------------
__global__ void __launch_bounds__(128)
layernorm(const float* __restrict__ X, const float* __restrict__ g,
          const float* __restrict__ bta, float* __restrict__ Y) {
  const int tid = threadIdx.x, wave = tid >> 5, lane = tid & 31;
  const size_t tok = (size_t)blockIdx.x * 4 + wave;
  const float* x = X + tok * E_DIM + lane * 8;

  float v[8];
#pragma unroll
  for (int i = 0; i < 8; ++i) v[i] = x[i];

  float s = 0.0f;
#pragma unroll
  for (int i = 0; i < 8; ++i) s += v[i];
  s += __shfl_xor(s, 1);  s += __shfl_xor(s, 2);
  s += __shfl_xor(s, 4);  s += __shfl_xor(s, 8);  s += __shfl_xor(s, 16);
  const float mean = s * (1.0f / E_DIM);

  float var = 0.0f;
#pragma unroll
  for (int i = 0; i < 8; ++i) { const float d = v[i] - mean; var += d * d; }
  var += __shfl_xor(var, 1);  var += __shfl_xor(var, 2);
  var += __shfl_xor(var, 4);  var += __shfl_xor(var, 8);  var += __shfl_xor(var, 16);
  const float inv = rsqrtf(var * (1.0f / E_DIM) + 1e-5f);

  float* y = Y + tok * E_DIM + lane * 8;
#pragma unroll
  for (int i = 0; i < 8; ++i) {
    const int e = lane * 8 + i;
    y[i] = (v[i] - mean) * inv * g[e] + bta[e];
  }
}

// ---------------------------------------------------------------------------
// Host-side launch sequence
// ---------------------------------------------------------------------------
extern "C" void kernel_launch(void* const* d_in, const int* in_sizes, int n_in,
                              void* d_out, int out_size, void* d_ws, size_t ws_size,
                              hipStream_t stream) {
  const float* query = (const float*)d_in[0];
  const float* value = (const float*)d_in[1];
  const float* qpos  = (const float*)d_in[2];
  const float* vpos  = (const float*)d_in[3];
  const float* Wqkv  = (const float*)d_in[4];
  const float* bqkv  = (const float*)d_in[5];
  const float* Wo    = (const float*)d_in[6];
  const float* bo    = (const float*)d_in[7];
  const float* ln1g  = (const float*)d_in[8];
  const float* ln1b  = (const float*)d_in[9];
  const float* W1    = (const float*)d_in[10];
  const float* b1    = (const float*)d_in[11];
  const float* W2    = (const float*)d_in[12];
  const float* b2    = (const float*)d_in[13];
  const float* ln2g  = (const float*)d_in[14];
  const float* ln2b  = (const float*)d_in[15];
  float* out = (float*)d_out;

  char* ws = (char*)d_ws;
  size_t off = 0;
  auto walloc = [&](size_t bytes) -> void* {
    void* p = ws + off;
    off += (bytes + 255) & ~(size_t)255;
    return p;
  };

  h16*   whqkv = (h16*)walloc((size_t)NL * 3 * E_DIM * E_DIM * sizeof(h16));
  h16*   who   = (h16*)walloc((size_t)NL * E_DIM * E_DIM * sizeof(h16));
  h16*   wh1   = (h16*)walloc((size_t)NL * E_DIM * E_DIM * sizeof(h16));
  h16*   wh2   = (h16*)walloc((size_t)NL * E_DIM * E_DIM * sizeof(h16));
  float* qpre  = (float*)walloc((size_t)NQ * E_DIM * sizeof(float));
  float* kvpre = (float*)walloc((size_t)NKV * E_DIM * sizeof(float));  // reused k then v
  h16*   qhb   = (h16*)walloc((size_t)NQ * E_DIM * sizeof(h16));
  h16*   khb   = (h16*)walloc((size_t)NKV * E_DIM * sizeof(h16));
  h16*   vhb   = (h16*)walloc((size_t)NKV * E_DIM * sizeof(h16));
  float* obuf  = (float*)walloc((size_t)NQ * E_DIM * sizeof(float));
  float* xbuf  = (float*)walloc((size_t)NQ * E_DIM * sizeof(float));
  float* hbuf  = (float*)walloc((size_t)NQ * E_DIM * sizeof(float));

  // convert all weights to f16 once
  {
    const int nqkv = NL * 3 * E_DIM * E_DIM;
    const int nsq  = NL * E_DIM * E_DIM;
    cvt_f32_f16<<<(nqkv + 255) / 256, 256, 0, stream>>>(Wqkv, whqkv, nqkv);
    cvt_f32_f16<<<(nsq + 255) / 256, 256, 0, stream>>>(Wo, who, nsq);
    cvt_f32_f16<<<(nsq + 255) / 256, 256, 0, stream>>>(W1, wh1, nsq);
    cvt_f32_f16<<<(nsq + 255) / 256, 256, 0, stream>>>(W2, wh2, nsq);
  }

  const float scale = 0.17677669529663687f;  // HD^-0.5
  const float* cur = query;

  for (int l = 0; l < NL; ++l) {
    const h16* Wq = whqkv + (size_t)l * 3 * E_DIM * E_DIM;
    const h16* Wk = Wq + (size_t)E_DIM * E_DIM;
    const h16* Wv = Wk + (size_t)E_DIM * E_DIM;
    const float* bq = bqkv + (size_t)l * 3 * E_DIM;
    const float* bk = bq + E_DIM;
    const float* bv = bk + E_DIM;

    // Q projection (scaled) + rotary -> qh
    gemm16<<<NQ / 16, 128, 0, stream>>>(cur, Wq, bq, nullptr, qpre, scale, 0);
    rotary_heads<<<NQ, 128, 0, stream>>>(qpre, qpos, qhb, LQ, 1);

    // K projection + rotary -> kh
    gemm16<<<NKV / 16, 128, 0, stream>>>(value, Wk, bk, nullptr, kvpre, 1.0f, 0);
    rotary_heads<<<NKV, 128, 0, stream>>>(kvpre, vpos, khb, LKV, 1);

    // V projection -> vh (reuse kvpre)
    gemm16<<<NKV / 16, 128, 0, stream>>>(value, Wv, bv, nullptr, kvpre, 1.0f, 0);
    rotary_heads<<<NKV, 128, 0, stream>>>(kvpre, vpos, vhb, LKV, 0);

    // flash attention -> obuf  [Lq,B,E]
    attn_fa<<<256, 128, 0, stream>>>(qhb, khb, vhb, obuf);

    // x = LN1( cur + o @ Wo^T + bo )
    gemm16<<<NQ / 16, 128, 0, stream>>>(obuf, who + (size_t)l * E_DIM * E_DIM,
                                        bo + (size_t)l * E_DIM, cur, xbuf, 1.0f, 0);
    layernorm<<<NQ / 4, 128, 0, stream>>>(xbuf, ln1g + (size_t)l * E_DIM,
                                          ln1b + (size_t)l * E_DIM, xbuf);

    // FFN: y = LN2( x + relu(x@W1^T+b1)@W2^T + b2 )
    gemm16<<<NQ / 16, 128, 0, stream>>>(xbuf, wh1 + (size_t)l * E_DIM * E_DIM,
                                        b1 + (size_t)l * E_DIM, nullptr, hbuf, 1.0f, 1);
    float* yout = out + (size_t)l * NQ * E_DIM;
    gemm16<<<NQ / 16, 128, 0, stream>>>(hbuf, wh2 + (size_t)l * E_DIM * E_DIM,
                                        b2 + (size_t)l * E_DIM, xbuf, yout, 1.0f, 0);
    layernorm<<<NQ / 4, 128, 0, stream>>>(yout, ln2g + (size_t)l * E_DIM,
                                          ln2b + (size_t)l * E_DIM, yout);

    cur = yout;  // next layer input
  }
}
